// UnifiedAdaptivePIDFusion_4458176053670
// MI455X (gfx1250) — matrix operand order repaired and checked by reading, not compiled, not run on hardware
//
#include <hip/hip_runtime.h>

typedef __attribute__((ext_vector_type(16))) __bf16 v16bf;
typedef __attribute__((ext_vector_type(8)))  __bf16 v8bf;
typedef __attribute__((ext_vector_type(4)))  __bf16 v4bf;
typedef __attribute__((ext_vector_type(8)))  float  v8f;

#define DD    768
#define NN    1024
#define KIN   832
#define BND   (16*1024*768)        // elems per output slot
#define W1SZ  (832*768)            // per-branch w1 elems
#define W2SZ  (768*768)            // per-branch w2 elems
#define XSTR  848                  // LDS row stride (halves) for x tile
#define HSTR  784                  // LDS row stride (halves) for h tile

__device__ __forceinline__ float geluf(float x) {
  return 0.5f * x * (1.0f + erff(x * 0.70710678118654752440f));
}
__device__ __forceinline__ float wave_sum32(float v) {
#pragma unroll
  for (int off = 1; off < 32; off <<= 1) v += __shfl_xor(v, off, 32);
  return v;
}

// ---------------------------------------------------------------- gmean ----
__global__ __launch_bounds__(256) void k_gmean(const float* __restrict__ gc,
                                               float* __restrict__ gmean) {
  const int b = blockIdx.x;
  const int d = blockIdx.y * 256 + threadIdx.x;
  const float* p = gc + (size_t)b * NN * DD + d;
  float s = 0.f;
#pragma unroll 4
  for (int n = 0; n < NN; ++n) s += p[(size_t)n * DD];
  gmean[b * DD + d] = s * (1.0f / NN);
}

// ---------------------------------------------------------------- cvt ------
__global__ __launch_bounds__(256) void k_cvt(const float* __restrict__ s,
                                             __bf16* __restrict__ d, int n) {
  int i = blockIdx.x * 256 + threadIdx.x;
  int stride = gridDim.x * 256;
  for (; i < n; i += stride) d[i] = (__bf16)s[i];
}

// ------------------------------------------------------------- small mlp ---
// one wave per batch row (16 rows, 512 threads)
__global__ __launch_bounds__(512) void k_smallmlp(
    const float* __restrict__ gmean, const float* __restrict__ qf,
    const int* __restrict__ mask,
    const float* __restrict__ gpw1, const float* __restrict__ gpb1,
    const float* __restrict__ gpg1, const float* __restrict__ gpbe1,
    const float* __restrict__ gpw2, const float* __restrict__ gpb2,
    const float* __restrict__ sw1p, const float* __restrict__ sb1p,
    const float* __restrict__ sg1p, const float* __restrict__ sbe1p,
    const float* __restrict__ sw2p, const float* __restrict__ sb2p,
    const float* __restrict__ sg2p, const float* __restrict__ sbe2p,
    const float* __restrict__ sw3p, const float* __restrict__ sb3p,
    float* __restrict__ gcc_ws, __bf16* __restrict__ ctx_ws,
    float* __restrict__ aw_out) {
  __shared__ float t1[16][384];
  __shared__ float gcs[16][64];
  __shared__ float h1[16][256];
  __shared__ float h2[16][128];
  __shared__ float lg[16][8];

  const int wv = threadIdx.x >> 5;   // batch row
  const int lane = threadIdx.x & 31;
  const float* gm = gmean + wv * DD;

  // ---- gcc = relu(ln(gmean @ gp_w1 + b)) @ gp_w2 + b ----
  float z[12];
  float ps = 0.f, pq = 0.f;
#pragma unroll 1
  for (int i = 0; i < 12; ++i) {
    const int j = lane + 32 * i;
    float s = gpb1[j];
#pragma unroll 4
    for (int dd = 0; dd < 768; ++dd) s += gm[dd] * gpw1[dd * 384 + j];
    z[i] = s; ps += s; pq += s * s;
  }
  ps = wave_sum32(ps); pq = wave_sum32(pq);
  float mean = ps * (1.f / 384.f);
  float rstd = rsqrtf(pq * (1.f / 384.f) - mean * mean + 1e-5f);
#pragma unroll
  for (int i = 0; i < 12; ++i) {
    const int j = lane + 32 * i;
    float v = (z[i] - mean) * rstd * gpg1[j] + gpbe1[j];
    t1[wv][j] = fmaxf(v, 0.f);
  }
  const float scale0 = mask[wv] ? 1.0f : 0.1f;
#pragma unroll 1
  for (int i = 0; i < 2; ++i) {
    const int j = lane + 32 * i;
    float s = gpb2[j];
#pragma unroll 4
    for (int dd = 0; dd < 384; ++dd) s += t1[wv][dd] * gpw2[dd * 64 + j];
    gcs[wv][j] = s;
    gcc_ws[wv * 64 + j] = s;
    ctx_ws[0 * 1024 + wv * 64 + j] = (__bf16)(s * scale0);  // ctx_s
    ctx_ws[1 * 1024 + wv * 64 + j] = (__bf16)s;             // ctx
  }
  // ---- swp layer1: wi=[gcc|qf] (832) -> 256, ln, gelu ----
  const float* qr = qf + wv * DD;
  ps = 0.f; pq = 0.f;
#pragma unroll 1
  for (int i = 0; i < 8; ++i) {
    const int j = lane + 32 * i;
    float s = sb1p[j];
#pragma unroll 4
    for (int dd = 0; dd < 64; ++dd) s += gcs[wv][dd] * sw1p[dd * 256 + j];
#pragma unroll 4
    for (int dd = 0; dd < 768; ++dd) s += qr[dd] * sw1p[(64 + dd) * 256 + j];
    z[i] = s; ps += s; pq += s * s;
  }
  ps = wave_sum32(ps); pq = wave_sum32(pq);
  mean = ps * (1.f / 256.f);
  rstd = rsqrtf(pq * (1.f / 256.f) - mean * mean + 1e-5f);
#pragma unroll
  for (int i = 0; i < 8; ++i) {
    const int j = lane + 32 * i;
    h1[wv][j] = geluf((z[i] - mean) * rstd * sg1p[j] + sbe1p[j]);
  }
  // ---- swp layer2: 256 -> 128, ln, gelu ----
  ps = 0.f; pq = 0.f;
#pragma unroll 1
  for (int i = 0; i < 4; ++i) {
    const int j = lane + 32 * i;
    float s = sb2p[j];
#pragma unroll 4
    for (int dd = 0; dd < 256; ++dd) s += h1[wv][dd] * sw2p[dd * 128 + j];
    z[i] = s; ps += s; pq += s * s;
  }
  ps = wave_sum32(ps); pq = wave_sum32(pq);
  mean = ps * (1.f / 128.f);
  rstd = rsqrtf(pq * (1.f / 128.f) - mean * mean + 1e-5f);
#pragma unroll
  for (int i = 0; i < 4; ++i) {
    const int j = lane + 32 * i;
    h2[wv][j] = geluf((z[i] - mean) * rstd * sg2p[j] + sbe2p[j]);
  }
  // ---- logits (8) + softmax ----
  if (lane < 8) {
    float s = sb3p[lane];
#pragma unroll 4
    for (int dd = 0; dd < 128; ++dd) s += h2[wv][dd] * sw3p[dd * 8 + lane];
    lg[wv][lane] = s;
  }
  if (lane == 0) {
    float mx = -1e30f;
    for (int j = 0; j < 8; ++j) mx = fmaxf(mx, lg[wv][j]);
    float e[8], sum = 0.f;
    for (int j = 0; j < 8; ++j) { e[j] = expf(lg[wv][j] - mx); sum += e[j]; }
    const float inv = 1.f / sum;
    for (int j = 0; j < 8; ++j) aw_out[wv * 8 + j] = e[j] * inv;
  }
}

// ------------------------------------------------------------- fused GEMM --
// grid = (512 tiles of 32 tokens, 8 branches), 256 threads (8 waves).
// Each wave: 2 M-tiles x 6 N-tiles (32x96 strip); each B fragment feeds 2 WMMAs.
__global__ __launch_bounds__(256) void k_fused(
    const float* __restrict__ comps, const __bf16* __restrict__ ctx_ws,
    const __bf16* __restrict__ w1ws, const __bf16* __restrict__ w2ws,
    const int* __restrict__ mask,
    const float* __restrict__ sb1, const float* __restrict__ sg1,
    const float* __restrict__ sbe1, const float* __restrict__ sb2,
    const float* __restrict__ sg2, const float* __restrict__ sbe2,
    const float* __restrict__ rb1, const float* __restrict__ rg1,
    const float* __restrict__ rbe1, const float* __restrict__ rb2,
    float* __restrict__ out) {
  const int k = blockIdx.y;
  const int tile = blockIdx.x;   // 32 tokens per tile
  const int tid = threadIdx.x;
  const int b = tile >> 5;       // 32 tiles per batch

  const int srcTab[8] = {0, 1, 3, 5, 4, 2, 6, 7};  // src comp == out slot
  const int srcC = srcTab[k];
  const bool isS = (k < 4);
  const int kk = isS ? k : (k - 4);
  const bool masked = (k >= 5);

  const float* src = comps + (size_t)srcC * BND + (size_t)tile * 32 * DD;
  float* dst = out + (size_t)srcC * BND + (size_t)tile * 32 * DD;

  if (masked && (mask[b] == 0)) {  // out = comps[src] verbatim
    const float4* s4 = (const float4*)src;
    float4* d4 = (float4*)dst;
    for (int i = tid; i < 6144; i += 256) d4[i] = s4[i];
    return;
  }

  // x tile (32x832, stride 848) and h tile (32x768, stride 784) alias the
  // same LDS: x is dead after GEMM1, and two barriers separate last x read
  // from first h write.
  __shared__ __align__(16) __bf16 smem[32 * XSTR];  // 54272 B
  __bf16* lx = smem;
  __bf16* lh = smem;
  __shared__ float part_s[8][32];
  __shared__ float part_q[8][32];
  __shared__ float rsum[32];
  __shared__ float rsq[32];

  // ---- stage x tile into LDS (f32 -> bf16) ----
  for (int i4 = tid; i4 < 6144; i4 += 256) {
    const int idx = i4 * 4;
    const int m = idx / DD;
    const int d = idx - m * DD;
    float4 v = *(const float4*)(src + idx);
    v4bf o;
    o[0] = (__bf16)v.x; o[1] = (__bf16)v.y; o[2] = (__bf16)v.z; o[3] = (__bf16)v.w;
    *(v4bf*)&lx[m * XSTR + d] = o;
  }
  const int variant = (k <= 4) ? 0 : 1;  // s-branches + r0 use scaled ctx
  for (int i = tid; i < 2048; i += 256) {
    const int m = i >> 6, j = i & 63;
    lx[m * XSTR + 768 + j] = ctx_ws[variant * 1024 + b * 64 + j];
  }
  __syncthreads();

  const int wv = tid >> 5, lane = tid & 31;
  const int hi = lane >> 4, nlo = lane & 15;
  const int n0w = wv * 96;  // each wave owns 32 x 96 of the output

  v8f acc[2][6];
#pragma unroll
  for (int mt = 0; mt < 2; ++mt)
#pragma unroll
    for (int t = 0; t < 6; ++t)
#pragma unroll
      for (int e = 0; e < 8; ++e) acc[mt][t][e] = 0.f;

  // ---- GEMM1: (32 x 832) @ (832 x 768) ----
  const __bf16* w1p = w1ws + (size_t)k * W1SZ;
#pragma unroll 1
  for (int kc = 0; kc < 26; ++kc) {
    const int kb = kc * 32;
    v16bf A[2];
#pragma unroll
    for (int mt = 0; mt < 2; ++mt) {
      const __bf16* ap = &lx[(mt * 16 + nlo) * XSTR + kb + hi * 8];
      v8bf alo = *(const v8bf*)ap;
      v8bf ah2 = *(const v8bf*)(ap + 16);
      A[mt] = __builtin_shufflevector(alo, ah2, 0, 1, 2, 3, 4, 5, 6, 7, 8, 9,
                                      10, 11, 12, 13, 14, 15);
    }
    const __bf16* bp = w1p + (size_t)(kb + hi * 8) * DD + n0w + nlo;
#pragma unroll
    for (int t = 0; t < 6; ++t) {
      v16bf Bm;
#pragma unroll
      for (int s = 0; s < 16; ++s)
        Bm[s] = bp[(size_t)(s < 8 ? s : s + 8) * DD + t * 16];
      acc[0][t] = __builtin_amdgcn_wmma_f32_16x16x32_bf16(
          false, A[0], false, Bm, (short)0, acc[0][t], false, false);
      acc[1][t] = __builtin_amdgcn_wmma_f32_16x16x32_bf16(
          false, A[1], false, Bm, (short)0, acc[1][t], false, false);
    }
  }

  // ---- bias + LayerNorm(768) + activation, h -> LDS bf16 ----
  const float* B1 = (isS ? sb1 : rb1) + kk * DD;
  const float* G1 = (isS ? sg1 : rg1) + kk * DD;
  const float* E1 = (isS ? sbe1 : rbe1) + kk * DD;
#pragma unroll
  for (int t = 0; t < 6; ++t) {
    const float bias = B1[n0w + t * 16 + nlo];
#pragma unroll
    for (int mt = 0; mt < 2; ++mt)
#pragma unroll
      for (int r = 0; r < 8; ++r) acc[mt][t][r] += bias;
  }
#pragma unroll
  for (int mt = 0; mt < 2; ++mt)
#pragma unroll
    for (int r = 0; r < 8; ++r) {
      float rv = 0.f, rq = 0.f;
#pragma unroll
      for (int t = 0; t < 6; ++t) {
        float v = acc[mt][t][r]; rv += v; rq += v * v;
      }
#pragma unroll
      for (int off = 1; off < 16; off <<= 1) {
        rv += __shfl_xor(rv, off, 32);
        rq += __shfl_xor(rq, off, 32);
      }
      if (nlo == 0) {
        part_s[wv][mt * 16 + r + 8 * hi] = rv;
        part_q[wv][mt * 16 + r + 8 * hi] = rq;
      }
    }
  __syncthreads();
  if (tid < 32) {
    float s = 0.f, q = 0.f;
#pragma unroll
    for (int w = 0; w < 8; ++w) { s += part_s[w][tid]; q += part_q[w][tid]; }
    rsum[tid] = s; rsq[tid] = q;
  }
  __syncthreads();
  float mu[2][8], rs[2][8];
#pragma unroll
  for (int mt = 0; mt < 2; ++mt)
#pragma unroll
    for (int r = 0; r < 8; ++r) {
      const int m = mt * 16 + r + 8 * hi;
      const float mn = rsum[m] * (1.f / 768.f);
      mu[mt][r] = mn;
      rs[mt][r] = rsqrtf(rsq[m] * (1.f / 768.f) - mn * mn + 1e-5f);
    }
#pragma unroll
  for (int t = 0; t < 6; ++t) {
    const int n = n0w + t * 16 + nlo;
    const float g = G1[n], e = E1[n];
#pragma unroll
    for (int mt = 0; mt < 2; ++mt)
#pragma unroll
      for (int r = 0; r < 8; ++r) {
        const float zz = (acc[mt][t][r] - mu[mt][r]) * rs[mt][r] * g + e;
        const float a = isS ? geluf(zz) : fmaxf(zz, 0.f);
        lh[(mt * 16 + r + 8 * hi) * HSTR + n] = (__bf16)a;
      }
  }
  __syncthreads();

  // ---- GEMM2: (32 x 768) @ (768 x 768) ----
#pragma unroll
  for (int mt = 0; mt < 2; ++mt)
#pragma unroll
    for (int t = 0; t < 6; ++t)
#pragma unroll
      for (int e = 0; e < 8; ++e) acc[mt][t][e] = 0.f;
  const __bf16* w2p = w2ws + (size_t)k * W2SZ;
#pragma unroll 1
  for (int kc = 0; kc < 24; ++kc) {
    const int kb = kc * 32;
    v16bf A[2];
#pragma unroll
    for (int mt = 0; mt < 2; ++mt) {
      const __bf16* ap = &lh[(mt * 16 + nlo) * HSTR + kb + hi * 8];
      v8bf alo = *(const v8bf*)ap;
      v8bf ah2 = *(const v8bf*)(ap + 16);
      A[mt] = __builtin_shufflevector(alo, ah2, 0, 1, 2, 3, 4, 5, 6, 7, 8, 9,
                                      10, 11, 12, 13, 14, 15);
    }
    const __bf16* bp = w2p + (size_t)(kb + hi * 8) * DD + n0w + nlo;
#pragma unroll
    for (int t = 0; t < 6; ++t) {
      v16bf Bm;
#pragma unroll
      for (int s = 0; s < 16; ++s)
        Bm[s] = bp[(size_t)(s < 8 ? s : s + 8) * DD + t * 16];
      acc[0][t] = __builtin_amdgcn_wmma_f32_16x16x32_bf16(
          false, A[0], false, Bm, (short)0, acc[0][t], false, false);
      acc[1][t] = __builtin_amdgcn_wmma_f32_16x16x32_bf16(
          false, A[1], false, Bm, (short)0, acc[1][t], false, false);
    }
  }

  // ---- bias (+ second LayerNorm for s-branches), store ----
  const float* B2 = (isS ? sb2 : rb2) + kk * DD;
#pragma unroll
  for (int t = 0; t < 6; ++t) {
    const float bias = B2[n0w + t * 16 + nlo];
#pragma unroll
    for (int mt = 0; mt < 2; ++mt)
#pragma unroll
      for (int r = 0; r < 8; ++r) acc[mt][t][r] += bias;
  }
  if (isS) {  // uniform across block -> barriers legal
#pragma unroll
    for (int mt = 0; mt < 2; ++mt)
#pragma unroll
      for (int r = 0; r < 8; ++r) {
        float rv = 0.f, rq = 0.f;
#pragma unroll
        for (int t = 0; t < 6; ++t) {
          float v = acc[mt][t][r]; rv += v; rq += v * v;
        }
#pragma unroll
        for (int off = 1; off < 16; off <<= 1) {
          rv += __shfl_xor(rv, off, 32);
          rq += __shfl_xor(rq, off, 32);
        }
        if (nlo == 0) {
          part_s[wv][mt * 16 + r + 8 * hi] = rv;
          part_q[wv][mt * 16 + r + 8 * hi] = rq;
        }
      }
    __syncthreads();
    if (tid < 32) {
      float s = 0.f, q = 0.f;
#pragma unroll
      for (int w = 0; w < 8; ++w) { s += part_s[w][tid]; q += part_q[w][tid]; }
      rsum[tid] = s; rsq[tid] = q;
    }
    __syncthreads();
    const float* G2 = sg2 + kk * DD;
    const float* E2 = sbe2 + kk * DD;
#pragma unroll
    for (int mt = 0; mt < 2; ++mt)
#pragma unroll
      for (int r = 0; r < 8; ++r) {
        const int m = mt * 16 + r + 8 * hi;
        const float mn = rsum[m] * (1.f / 768.f);
        mu[mt][r] = mn;
        rs[mt][r] = rsqrtf(rsq[m] * (1.f / 768.f) - mn * mn + 1e-5f);
      }
#pragma unroll
    for (int t = 0; t < 6; ++t) {
      const int n = n0w + t * 16 + nlo;
      const float g = G2[n], e = E2[n];
#pragma unroll
      for (int mt = 0; mt < 2; ++mt)
#pragma unroll
        for (int r = 0; r < 8; ++r)
          acc[mt][t][r] = (acc[mt][t][r] - mu[mt][r]) * rs[mt][r] * g + e;
    }
  }
#pragma unroll
  for (int t = 0; t < 6; ++t) {
    const int n = n0w + t * 16 + nlo;
#pragma unroll
    for (int mt = 0; mt < 2; ++mt)
#pragma unroll
      for (int r = 0; r < 8; ++r) {
        const int m = mt * 16 + r + 8 * hi;
        dst[(size_t)m * DD + n] = acc[mt][t][r];
      }
  }
}

// --------------------------------------------------------------- launch ----
extern "C" void kernel_launch(void* const* d_in, const int* in_sizes, int n_in,
                              void* d_out, int out_size, void* d_ws,
                              size_t ws_size, hipStream_t stream) {
  (void)in_sizes; (void)n_in; (void)out_size; (void)ws_size;
  const float* comps = (const float*)d_in[0];
  const float* gc    = (const float*)d_in[1];
  const float* qf    = (const float*)d_in[2];
  const int*   mask  = (const int*)d_in[3];
  const float* gpw1  = (const float*)d_in[4];
  const float* gpb1  = (const float*)d_in[5];
  const float* gpg1  = (const float*)d_in[6];
  const float* gpbe1 = (const float*)d_in[7];
  const float* gpw2  = (const float*)d_in[8];
  const float* gpb2  = (const float*)d_in[9];
  const float* swpw1 = (const float*)d_in[10];
  const float* swpb1 = (const float*)d_in[11];
  const float* swpg1 = (const float*)d_in[12];
  const float* swpbe1= (const float*)d_in[13];
  const float* swpw2 = (const float*)d_in[14];
  const float* swpb2 = (const float*)d_in[15];
  const float* swpg2 = (const float*)d_in[16];
  const float* swpbe2= (const float*)d_in[17];
  const float* swpw3 = (const float*)d_in[18];
  const float* swpb3 = (const float*)d_in[19];
  const float* sw1   = (const float*)d_in[20];
  const float* sb1   = (const float*)d_in[21];
  const float* sg1   = (const float*)d_in[22];
  const float* sbe1  = (const float*)d_in[23];
  const float* sw2   = (const float*)d_in[24];
  const float* sb2   = (const float*)d_in[25];
  const float* sg2   = (const float*)d_in[26];
  const float* sbe2  = (const float*)d_in[27];
  const float* rw1   = (const float*)d_in[28];
  const float* rb1   = (const float*)d_in[29];
  const float* rg1   = (const float*)d_in[30];
  const float* rbe1  = (const float*)d_in[31];
  const float* rw2   = (const float*)d_in[32];
  const float* rb2   = (const float*)d_in[33];

  char* ws = (char*)d_ws;
  float*  gmean = (float*)(ws + 0);                    // 16x768 f32
  float*  gccw  = (float*)(ws + 49152);                // 16x64  f32
  __bf16* ctx   = (__bf16*)(ws + 53248);               // 2x16x64 bf16
  __bf16* w1    = (__bf16*)(ws + 57344);               // 8x832x768 bf16
  __bf16* w2    = (__bf16*)(ws + 57344 + 10223616);    // 8x768x768 bf16

  float* outp = (float*)d_out;
  float* aw = outp + (size_t)8 * BND;  // aw tail: 16x8

  k_gmean<<<dim3(16, 3), 256, 0, stream>>>(gc, gmean);
  k_smallmlp<<<1, 512, 0, stream>>>(
      gmean, qf, mask, gpw1, gpb1, gpg1, gpbe1, gpw2, gpb2, swpw1, swpb1,
      swpg1, swpbe1, swpw2, swpb2, swpg2, swpbe2, swpw3, swpb3, gccw, ctx, aw);
  k_cvt<<<2048, 256, 0, stream>>>(sw1, w1, 4 * W1SZ);
  k_cvt<<<2048, 256, 0, stream>>>(rw1, w1 + 4 * W1SZ, 4 * W1SZ);
  k_cvt<<<2048, 256, 0, stream>>>(sw2, w2, 4 * W2SZ);
  k_cvt<<<2048, 256, 0, stream>>>(rw2, w2 + 4 * W2SZ, 4 * W2SZ);
  k_fused<<<dim3(512, 8), 256, 0, stream>>>(
      comps, ctx, w1, w2, mask, sb1, sg1, sbe1, sb2, sg2, sbe2, rb1, rg1,
      rbe1, rb2, outp);
}
